// NMT_60662118088983
// MI455X (gfx1250) — compile-verified, matrix operand order checked by reading, not run on hardware
//
#include <hip/hip_runtime.h>
#include <math.h>

typedef __attribute__((ext_vector_type(16))) _Float16 v16h;
typedef __attribute__((ext_vector_type(8)))  float    v8f;

#define SRC_LEN 50
#define TRG_LEN 40
#define BATCH   128
#define SRC_V   32000
#define TRG_V   30000
#define EE      30
#define EH      64
#define DH      128
#define NSTEP   39                       // TRG_LEN - 1 decode steps
#define OUT_SLICE ((size_t)BATCH * TRG_V)  // 3,840,000 floats per output row-slice

// ---------------------------------------------------------------------------
// WMMA fragment helpers (CDNA5 16x16x32 f16 layout, wave32)
// A (MxK) and B stored as W[N][K]: lane<16 -> k {0..7,16..23}, lane>=16 -> +8.
// ---------------------------------------------------------------------------
static __device__ __forceinline__ v16h load_frag(const _Float16* __restrict__ p,
                                                 int ld, int row0, int k0, int lane) {
  int r  = row0 + (lane & 15);
  int hf = (lane & 16) >> 1;                     // 0 or 8
  const _Float16* q = p + (size_t)r * ld + k0;
  union { uint4 u; _Float16 h[8]; } a, b;
  a.u = *(const uint4*)(q + hf);
  b.u = *(const uint4*)(q + 16 + hf);
  v16h v;
#pragma unroll
  for (int j = 0; j < 8; ++j) { v[j] = a.h[j]; v[8 + j] = b.h[j]; }
  return v;
}

static __device__ __forceinline__ v8f wmma_f16(v16h a, v16h b, v8f c) {
  return __builtin_amdgcn_wmma_f32_16x16x32_f16(false, a, false, b, (short)0, c,
                                                false, false);
}

static __device__ __forceinline__ float sigmoidf_(float x) {
  return 1.0f / (1.0f + __expf(-x));
}

// ---------------------------------------------------------------------------
// prep: weight f16 conversion, input projections X = emb @ w_ih^T + b,
//       output row0, accumulator zeroing (every call -> replay-deterministic)
// ---------------------------------------------------------------------------
__global__ void prep_kernel(
    const int* __restrict__ src, const float* __restrict__ EEMB,
    const float* __restrict__ ewih_f, const float* __restrict__ eb_f,
    const float* __restrict__ ewih_b, const float* __restrict__ eb_b,
    const float* __restrict__ ewhh_f, const float* __restrict__ ewhh_b,
    const float* __restrict__ attnWi, const float* __restrict__ attnWo,
    const float* __restrict__ dwih,   const float* __restrict__ dwhh,
    const float* __restrict__ genW,
    float* __restrict__ out,
    _Float16* __restrict__ genW_h,   _Float16* __restrict__ attnWi_h,
    _Float16* __restrict__ attnWo_h, _Float16* __restrict__ decWih_h,
    _Float16* __restrict__ decWhh_h, _Float16* __restrict__ encWhh_h,
    float* __restrict__ X, float* __restrict__ sumexp,
    unsigned long long* __restrict__ keys)
{
  const size_t gid    = (size_t)blockIdx.x * blockDim.x + threadIdx.x;
  const size_t stride = (size_t)gridDim.x * blockDim.x;

  // output row 0: -1 everywhere, 0 at vocab index 2
  for (size_t i = gid; i < OUT_SLICE; i += stride)
    out[i] = ((i % TRG_V) == 2) ? 0.0f : -1.0f;
  // gen_W -> f16 (stays L2-resident: 7.7 MB)
  for (size_t i = gid; i < (size_t)TRG_V * DH; i += stride)
    genW_h[i] = (_Float16)genW[i];
  for (size_t i = gid; i < 128u * 128u; i += stride) attnWi_h[i] = (_Float16)attnWi[i];
  for (size_t i = gid; i < 128u * 256u; i += stride) attnWo_h[i] = (_Float16)attnWo[i];
  // dec_w_ih [512][158] -> padded [512][160]
  for (size_t i = gid; i < 512u * 160u; i += stride) {
    int n = (int)(i / 160), k = (int)(i % 160);
    decWih_h[i] = (_Float16)((k < 158) ? dwih[n * 158 + k] : 0.0f);
  }
  for (size_t i = gid; i < 512u * 128u; i += stride) decWhh_h[i] = (_Float16)dwhh[i];
  for (size_t i = gid; i < 256u * 64u; i += stride) {
    encWhh_h[i]            = (_Float16)ewhh_f[i];
    encWhh_h[256 * 64 + i] = (_Float16)ewhh_b[i];
  }
  // X[dir][s][b][j] = b[j] + sum_e EEMB[tok][e] * w_ih[j][e]
  const size_t NX = (size_t)2 * SRC_LEN * BATCH * 256;
  for (size_t i = gid; i < NX; i += stride) {
    size_t r = i;
    int dir = (int)(r / ((size_t)SRC_LEN * BATCH * 256)); r %= (size_t)SRC_LEN * BATCH * 256;
    int s   = (int)(r / (BATCH * 256));                   r %= (size_t)BATCH * 256;
    int b   = (int)(r / 256);
    int j   = (int)(r % 256);
    int tok = src[s * BATCH + b];
    const float* wi = dir ? ewih_b : ewih_f;
    float acc = dir ? eb_b[j] : eb_f[j];
    const float* e = EEMB + (size_t)tok * EE;
    const float* w = wi + j * EE;
#pragma unroll 6
    for (int k = 0; k < EE; ++k) acc += e[k] * w[k];
    X[i] = acc;
  }
  for (size_t i = gid; i < (size_t)NSTEP * BATCH; i += stride) {
    sumexp[i] = 0.0f;
    keys[i]   = 0ull;
  }
}

// ---------------------------------------------------------------------------
// Encoder: bidirectional LSTM, one block per direction, WMMA recurrence
//   g = X[s] + h @ w_hh^T ; cell ; write enc_out; final interleave to dec state
// ---------------------------------------------------------------------------
__global__ __launch_bounds__(1024) void enc_kernel(
    const float* __restrict__ X,            // [2][50][128][256]
    const _Float16* __restrict__ Whh_h,     // [2][256][64]
    float* __restrict__ hbuf, float* __restrict__ cbuf,   // [2][128][64]
    _Float16* __restrict__ h16buf,                        // [2][128][64]
    float* __restrict__ gbuf,                             // [2][128][256]
    float* __restrict__ enc_out,                          // [50][128][128]
    float* __restrict__ dec_h, float* __restrict__ dec_c, // [128][128]
    _Float16* __restrict__ dec_h16)                       // [128][128]
{
  const int dir  = blockIdx.x;
  const int tid  = threadIdx.x;
  const int lane = tid & 31;
  const int wv   = tid >> 5;
  const float*    Xd  = X + (size_t)dir * SRC_LEN * BATCH * 256;
  const _Float16* W   = Whh_h + dir * 256 * 64;
  float*    h   = hbuf   + dir * BATCH * EH;
  float*    c   = cbuf   + dir * BATCH * EH;
  _Float16* h16 = h16buf + dir * BATCH * EH;
  float*    g   = gbuf   + dir * BATCH * 256;

  for (int idx = tid; idx < BATCH * EH; idx += 1024) {
    h[idx] = 0.0f; c[idx] = 0.0f; h16[idx] = (_Float16)0.0f;
  }
  __threadfence(); __syncthreads();

  for (int step = 0; step < SRC_LEN; ++step) {
    int s = dir ? (SRC_LEN - 1 - step) : step;
    const float* Xs = Xd + (size_t)s * BATCH * 256;
    // M=128 (8 tiles) x N=256 (16 tiles), K=64 (2 chunks): 4 tiles / wave
#pragma unroll
    for (int rep = 0; rep < 4; ++rep) {
      int idx = wv + 32 * rep;
      int mt = idx >> 4, nt = idx & 15;
      v8f acc = {};
      acc = wmma_f16(load_frag(h16, 64, mt * 16, 0,  lane),
                     load_frag(W,   64, nt * 16, 0,  lane), acc);
      acc = wmma_f16(load_frag(h16, 64, mt * 16, 32, lane),
                     load_frag(W,   64, nt * 16, 32, lane), acc);
      int col  = nt * 16 + (lane & 15);
      int rbse = mt * 16 + ((lane & 16) >> 1);
#pragma unroll
      for (int i = 0; i < 8; ++i)
        g[(rbse + i) * 256 + col] = acc[i] + Xs[(rbse + i) * 256 + col];
    }
    __threadfence(); __syncthreads();
    for (int idx = tid; idx < BATCH * EH; idx += 1024) {
      int b = idx >> 6, d = idx & 63;
      const float* gb = g + b * 256;
      float gi = gb[d], gf = gb[64 + d], gg = gb[128 + d], go = gb[192 + d];
      float cn = sigmoidf_(gf) * c[idx] + sigmoidf_(gi) * tanhf(gg);
      float hn = sigmoidf_(go) * tanhf(cn);
      c[idx] = cn; h[idx] = hn; h16[idx] = (_Float16)hn;
      enc_out[((size_t)s * BATCH + b) * 128 + dir * EH + d] = hn;
    }
    __threadfence(); __syncthreads();
  }
  // hidden/context = stack([hf,hb],-1).reshape -> interleave even/odd
  for (int idx = tid; idx < BATCH * EH; idx += 1024) {
    int b = idx >> 6, d = idx & 63;
    float hv = h[idx], cv = c[idx];
    dec_h[b * 128 + 2 * d + dir]   = hv;
    dec_c[b * 128 + 2 * d + dir]   = cv;
    dec_h16[b * 128 + 2 * d + dir] = (_Float16)hv;
  }
}

// ---------------------------------------------------------------------------
// Decoder step (1 block, 32 waves): attention + LSTM cell, all GEMMs via WMMA
// ---------------------------------------------------------------------------
__global__ __launch_bounds__(1024) void dec_step_kernel(
    const float* __restrict__ enc_out,        // [50][128][128]
    const _Float16* __restrict__ Wi_h,        // [128][128]
    const _Float16* __restrict__ Wo_h,        // [128][256]
    const _Float16* __restrict__ Wih_h,       // [512][160]
    const _Float16* __restrict__ Whh_h,       // [512][128]
    const float* __restrict__ dec_b,          // [512]
    const float* __restrict__ DEMB,           // [30000][30]
    const int* __restrict__ trg,              // [40][128]
    const int* __restrict__ training,
    const unsigned long long* __restrict__ keysPrev,  // [128]
    float* __restrict__ h, float* __restrict__ c,     // [128][128]
    _Float16* __restrict__ h16,                       // [128][128]
    float* __restrict__ semi,                         // [128][128]
    float* __restrict__ score,                        // [50][128]
    _Float16* __restrict__ cat16,                     // [128][256]
    _Float16* __restrict__ din16,                     // [128][160]
    float* __restrict__ g,                            // [128][512]
    int t)
{
  const int tid  = threadIdx.x;
  const int lane = tid & 31;
  const int wv   = tid >> 5;

  // A) semi = h @ Wi^T  (M=N=K=128)
#pragma unroll
  for (int rep = 0; rep < 2; ++rep) {
    int idx = wv + 32 * rep;
    int mt = idx >> 3, nt = idx & 7;
    v8f acc = {};
#pragma unroll
    for (int kc = 0; kc < 4; ++kc)
      acc = wmma_f16(load_frag(h16,  128, mt * 16, kc * 32, lane),
                     load_frag(Wi_h, 128, nt * 16, kc * 32, lane), acc);
    int col = nt * 16 + (lane & 15);
    int rb  = mt * 16 + ((lane & 16) >> 1);
#pragma unroll
    for (int i = 0; i < 8; ++i) semi[(rb + i) * 128 + col] = acc[i];
  }
  __threadfence(); __syncthreads();

  // B) raw attention scores + softmax over s
  for (int idx = tid; idx < SRC_LEN * BATCH; idx += 1024) {
    int s = idx / BATCH, b = idx % BATCH;
    const float* e  = enc_out + ((size_t)s * BATCH + b) * 128;
    const float* sm = semi + b * 128;
    float acc = 0.0f;
#pragma unroll 8
    for (int d = 0; d < 128; ++d) acc += e[d] * sm[d];
    score[s * BATCH + b] = acc;
  }
  __threadfence(); __syncthreads();
  if (tid < BATCH) {
    int b = tid;
    float mx = -INFINITY;
    for (int s = 0; s < SRC_LEN; ++s) mx = fmaxf(mx, score[s * BATCH + b]);
    float sum = 0.0f;
    for (int s = 0; s < SRC_LEN; ++s) sum += __expf(score[s * BATCH + b] - mx);
    float inv = 1.0f / sum;
    for (int s = 0; s < SRC_LEN; ++s)
      score[s * BATCH + b] = __expf(score[s * BATCH + b] - mx) * inv;
  }
  __threadfence(); __syncthreads();

  // C) s_tilde -> cat16[:, :128], h -> cat16[:, 128:]
  for (int idx = tid; idx < BATCH * 128; idx += 1024) {
    int b = idx >> 7, d = idx & 127;
    float acc = 0.0f;
    for (int s = 0; s < SRC_LEN; ++s)
      acc += score[s * BATCH + b] * enc_out[((size_t)s * BATCH + b) * 128 + d];
    cat16[b * 256 + d]       = (_Float16)acc;
    cat16[b * 256 + 128 + d] = h16[b * 128 + d];
  }
  __threadfence(); __syncthreads();

  // D) c_t = tanh(cat @ Wo^T) -> din16[:, :128]   (M=128,N=128,K=256)
#pragma unroll
  for (int rep = 0; rep < 2; ++rep) {
    int idx = wv + 32 * rep;
    int mt = idx >> 3, nt = idx & 7;
    v8f acc = {};
#pragma unroll
    for (int kc = 0; kc < 8; ++kc)
      acc = wmma_f16(load_frag(cat16, 256, mt * 16, kc * 32, lane),
                     load_frag(Wo_h,  256, nt * 16, kc * 32, lane), acc);
    int col = nt * 16 + (lane & 15);
    int rb  = mt * 16 + ((lane & 16) >> 1);
#pragma unroll
    for (int i = 0; i < 8; ++i)
      din16[(rb + i) * 160 + col] = (_Float16)tanhf(acc[i]);
  }
  __threadfence(); __syncthreads();

  // E) token embedding -> din16[:, 128:160] (teacher forcing or argmax word)
  for (int idx = tid; idx < BATCH * 32; idx += 1024) {
    int b = idx >> 5, j = idx & 31;
    int tok;
    if (training[0] != 0)  tok = trg[t * BATCH + b];
    else if (t == 0)       tok = 2;
    else {
      unsigned long long k = keysPrev[b];
      tok = 0x7FFFFFFF - (int)(unsigned)(k & 0xFFFFFFFFull);
    }
    float v = (j < EE) ? DEMB[(size_t)tok * EE + j] : 0.0f;
    din16[b * 160 + 128 + j] = (_Float16)v;
  }
  __threadfence(); __syncthreads();

  // F) gates g = din @ Wih^T + h @ Whh^T + b  (M=128, N=512, K=160/128)
  {
    int n0 = wv * 16;                        // each wave owns one N-tile of 512
#pragma unroll
    for (int mt = 0; mt < 8; ++mt) {
      v8f acc = {};
#pragma unroll
      for (int kc = 0; kc < 5; ++kc)
        acc = wmma_f16(load_frag(din16, 160, mt * 16, kc * 32, lane),
                       load_frag(Wih_h, 160, n0,      kc * 32, lane), acc);
#pragma unroll
      for (int kc = 0; kc < 4; ++kc)
        acc = wmma_f16(load_frag(h16,   128, mt * 16, kc * 32, lane),
                       load_frag(Whh_h, 128, n0,      kc * 32, lane), acc);
      int col   = n0 + (lane & 15);
      float bia = dec_b[col];
      int rb    = mt * 16 + ((lane & 16) >> 1);
#pragma unroll
      for (int i = 0; i < 8; ++i) g[(rb + i) * 512 + col] = acc[i] + bia;
    }
  }
  __threadfence(); __syncthreads();

  // G) LSTM cell update, emit new h (f32 + f16 for generator)
  for (int idx = tid; idx < BATCH * DH; idx += 1024) {
    int b = idx >> 7, d = idx & 127;
    const float* gb = g + b * 512;
    float gi = gb[d], gf = gb[128 + d], gg = gb[256 + d], go = gb[384 + d];
    float cn = sigmoidf_(gf) * c[idx] + sigmoidf_(gi) * tanhf(gg);
    float hn = sigmoidf_(go) * tanhf(cn);
    c[idx] = cn; h[idx] = hn; h16[idx] = (_Float16)hn;
  }
}

// ---------------------------------------------------------------------------
// Generator: z = h @ gen_W^T + b, write raw logits, accumulate sum(exp) and
// argmax via shuffle reduction + atomics. 125 blocks x 240 vocab cols.
// ---------------------------------------------------------------------------
__global__ __launch_bounds__(256) void gen_kernel(
    const _Float16* __restrict__ h16,      // [128][128]
    const _Float16* __restrict__ genW_h,   // [30000][128]
    const float* __restrict__ genb,        // [30000]
    float* __restrict__ outz,              // [128][30000]
    float* __restrict__ sumexp,            // [128]
    unsigned long long* __restrict__ amax) // [128]
{
  const int lane = threadIdx.x & 31;
  const int wv   = threadIdx.x >> 5;       // wave = M-tile (8 waves)
  const int m0   = wv * 16;
  const int rb   = (lane & 16) >> 1;

  v16h a0 = load_frag(h16, 128, m0, 0,  lane);
  v16h a1 = load_frag(h16, 128, m0, 32, lane);
  v16h a2 = load_frag(h16, 128, m0, 64, lane);
  v16h a3 = load_frag(h16, 128, m0, 96, lane);

  float rsum[8]; float rmax[8]; int rcol[8];
#pragma unroll
  for (int i = 0; i < 8; ++i) { rsum[i] = 0.0f; rmax[i] = -INFINITY; rcol[i] = 0x7FFFFFFF; }

  for (int nt = 0; nt < 15; ++nt) {
    int n0 = blockIdx.x * 240 + nt * 16;
    v8f acc = {};
    acc = wmma_f16(a0, load_frag(genW_h, 128, n0, 0,  lane), acc);
    acc = wmma_f16(a1, load_frag(genW_h, 128, n0, 32, lane), acc);
    acc = wmma_f16(a2, load_frag(genW_h, 128, n0, 64, lane), acc);
    acc = wmma_f16(a3, load_frag(genW_h, 128, n0, 96, lane), acc);
    int col    = n0 + (lane & 15);
    float bias = genb[col];
#pragma unroll
    for (int i = 0; i < 8; ++i) {
      float z = acc[i] + bias;
      int b   = m0 + rb + i;
      outz[(size_t)b * TRG_V + col] = z;
      rsum[i] += __expf(z);
      if (z > rmax[i] || (z == rmax[i] && col < rcol[i])) { rmax[i] = z; rcol[i] = col; }
    }
  }

#pragma unroll
  for (int i = 0; i < 8; ++i) {
    float s = rsum[i];
    float mx = rmax[i]; int cidx = rcol[i];
#pragma unroll
    for (int m = 8; m >= 1; m >>= 1) {
      s += __shfl_xor(s, m);
      float om = __shfl_xor(mx, m);
      int   oc = __shfl_xor(cidx, m);
      if (om > mx || (om == mx && oc < cidx)) { mx = om; cidx = oc; }
    }
    if ((lane & 15) == 0) {
      int b = m0 + rb + i;
      atomicAdd(&sumexp[b], s);
      unsigned u = __float_as_uint(mx);
      u = (u & 0x80000000u) ? ~u : (u | 0x80000000u);
      unsigned long long key =
          ((unsigned long long)u << 32) | (unsigned)(0x7FFFFFFF - cidx);
      atomicMax(amax + b, key);
    }
  }
}

// ---------------------------------------------------------------------------
// Normalize: lp = z - log(sum exp z), in place on the (L2-hot) output slice
// grid = 512 blocks: b = blk>>2, quarter = blk&3, float4-vectorized
// ---------------------------------------------------------------------------
__global__ __launch_bounds__(256) void norm_kernel(float* __restrict__ outz,
                                                   const float* __restrict__ sumexp)
{
  int b = blockIdx.x >> 2;
  int q = blockIdx.x & 3;
  float lse = __logf(sumexp[b]);
  float4* row = (float4*)(outz + (size_t)b * TRG_V);
  for (int j = q * 1875 + (int)threadIdx.x; j < (q + 1) * 1875; j += 256) {
    float4 v = row[j];
    v.x -= lse; v.y -= lse; v.z -= lse; v.w -= lse;
    row[j] = v;
  }
}

// ---------------------------------------------------------------------------
extern "C" void kernel_launch(void* const* d_in, const int* in_sizes, int n_in,
                              void* d_out, int out_size, void* d_ws, size_t ws_size,
                              hipStream_t stream) {
  (void)in_sizes; (void)n_in; (void)out_size; (void)ws_size;
  const int*   src      = (const int*)d_in[0];
  const int*   trg      = (const int*)d_in[1];
  const int*   training = (const int*)d_in[2];
  const float* EEMB     = (const float*)d_in[3];
  const float* DEMB     = (const float*)d_in[4];
  const float* ewih_f   = (const float*)d_in[5];
  const float* ewhh_f   = (const float*)d_in[6];
  const float* eb_f     = (const float*)d_in[7];
  const float* ewih_b   = (const float*)d_in[8];
  const float* ewhh_b   = (const float*)d_in[9];
  const float* eb_b     = (const float*)d_in[10];
  const float* attnWi   = (const float*)d_in[11];
  const float* attnWo   = (const float*)d_in[12];
  const float* dwih     = (const float*)d_in[13];
  const float* dwhh     = (const float*)d_in[14];
  const float* db       = (const float*)d_in[15];
  const float* genW     = (const float*)d_in[16];
  const float* genb     = (const float*)d_in[17];
  float* out = (float*)d_out;

  // workspace carve-up (256B aligned)
  char* ws = (char*)d_ws;
  size_t off = 0;
  auto take = [&](size_t bytes) -> char* {
    char* p = ws + off;
    off = (off + bytes + 255) & ~(size_t)255;
    return p;
  };
  _Float16* genW_h   = (_Float16*)take((size_t)TRG_V * DH * 2);
  _Float16* attnWi_h = (_Float16*)take(128 * 128 * 2);
  _Float16* attnWo_h = (_Float16*)take(128 * 256 * 2);
  _Float16* decWih_h = (_Float16*)take(512 * 160 * 2);
  _Float16* decWhh_h = (_Float16*)take(512 * 128 * 2);
  _Float16* encWhh_h = (_Float16*)take(2 * 256 * 64 * 2);
  float*    X        = (float*)take((size_t)2 * SRC_LEN * BATCH * 256 * 4);
  float*    enc_out  = (float*)take((size_t)SRC_LEN * BATCH * 128 * 4);
  float*    enc_h    = (float*)take(2 * BATCH * EH * 4);
  float*    enc_c    = (float*)take(2 * BATCH * EH * 4);
  _Float16* enc_h16  = (_Float16*)take(2 * BATCH * EH * 2);
  float*    enc_g    = (float*)take(2 * BATCH * 256 * 4);
  float*    dec_h    = (float*)take(BATCH * DH * 4);
  float*    dec_c    = (float*)take(BATCH * DH * 4);
  _Float16* dec_h16  = (_Float16*)take(BATCH * DH * 2);
  float*    semi     = (float*)take(BATCH * 128 * 4);
  float*    score    = (float*)take(SRC_LEN * BATCH * 4);
  _Float16* cat16    = (_Float16*)take(BATCH * 256 * 2);
  _Float16* din16    = (_Float16*)take(BATCH * 160 * 2);
  float*    gbuf     = (float*)take(BATCH * 512 * 4);
  float*    sumexp   = (float*)take((size_t)NSTEP * BATCH * 4);
  unsigned long long* keys = (unsigned long long*)take((size_t)NSTEP * BATCH * 8);

  prep_kernel<<<1024, 256, 0, stream>>>(
      src, EEMB, ewih_f, eb_f, ewih_b, eb_b, ewhh_f, ewhh_b,
      attnWi, attnWo, dwih, dwhh, genW, out,
      genW_h, attnWi_h, attnWo_h, decWih_h, decWhh_h, encWhh_h,
      X, sumexp, keys);

  enc_kernel<<<2, 1024, 0, stream>>>(
      X, encWhh_h, enc_h, enc_c, enc_h16, enc_g, enc_out,
      dec_h, dec_c, dec_h16);

  for (int t = 0; t < NSTEP; ++t) {
    const unsigned long long* kp = keys + (size_t)(t > 0 ? t - 1 : 0) * BATCH;
    dec_step_kernel<<<1, 1024, 0, stream>>>(
        enc_out, attnWi_h, attnWo_h, decWih_h, decWhh_h, db, DEMB,
        trg, training, kp, dec_h, dec_c, dec_h16,
        semi, score, cat16, din16, gbuf, t);
    float* outz = out + (size_t)(t + 1) * OUT_SLICE;
    gen_kernel<<<125, 256, 0, stream>>>(
        dec_h16, genW_h, genb, outz,
        sumexp + (size_t)t * BATCH, keys + (size_t)t * BATCH);
    norm_kernel<<<512, 256, 0, stream>>>(outz, sumexp + (size_t)t * BATCH);
  }
}
